// GNNEncoder_18743237280721
// MI455X (gfx1250) — matrix operand order repaired and checked by reading, not compiled, not run on hardware
//
#include <hip/hip_runtime.h>
#include <hip/hip_bf16.h>

#define N_NODES  100000
#define N_EDGES  1600000
#define DIM      128
#define N_GRAPHS 256

typedef __attribute__((ext_vector_type(16))) __bf16 v16bf;
typedef __attribute__((ext_vector_type(8)))  __bf16 v8bf;
typedef __attribute__((ext_vector_type(4)))  __bf16 v4bf;
typedef __attribute__((ext_vector_type(8)))  float  v8f;

__device__ __forceinline__ void atomic_fadd(float* p, float v) {
    // lowers to global_atomic_add_f32 (device scope) on gfx1250
    __hip_atomic_fetch_add(p, v, __ATOMIC_RELAXED, __HIP_MEMORY_SCOPE_AGENT);
}

// ---------------- elementwise helpers ----------------

__global__ void cvt_f32_bf16_kernel(const float* __restrict__ in,
                                    __bf16* __restrict__ out, int n) {
    int i = blockIdx.x * blockDim.x + threadIdx.x;
    if (i < n) out[i] = (__bf16)in[i];
}

__global__ void zero_f32_kernel(float* __restrict__ p, int n) {
    int i = blockIdx.x * blockDim.x + threadIdx.x;
    if (i < n) p[i] = 0.0f;
}

// ---------------- edge scatter (mean-aggregation numerator + degree) ----------------
// one wave32 per edge; lane handles 4 consecutive features (4x f32 atomic add).
__global__ void edge_scatter_kernel(const __bf16* __restrict__ feat,
                                    const int* __restrict__ src,
                                    const int* __restrict__ dst,
                                    float* __restrict__ msg,
                                    float* __restrict__ deg) {
    int gid  = blockIdx.x * blockDim.x + threadIdx.x;
    int e    = gid >> 5;
    int lane = gid & 31;
    if (e >= N_EDGES) return;
    int s = src[e];
    int d = dst[e];
    if ((unsigned)s >= N_NODES || (unsigned)d >= N_NODES) return;
    const __bf16* fp = feat + (size_t)s * DIM + lane * 4;
    float*        mp = msg  + (size_t)d * DIM + lane * 4;
    v4bf v = *(const v4bf*)fp;
    atomic_fadd(mp + 0, (float)v[0]);
    atomic_fadd(mp + 1, (float)v[1]);
    atomic_fadd(mp + 2, (float)v[2]);
    atomic_fadd(mp + 3, (float)v[3]);
    if (lane == 0) atomic_fadd(deg + d, 1.0f);
}

// ---------------- WMMA SAGE layer ----------------
// out = relu( (msg/max(deg,1)) @ Wl^T + b + xin @ Wr^T ), msg/xin tiles staged through LDS.
// Block = 256 threads = 8 waves. Block owns a 16-row (M) stripe of nodes; wave w
// owns N-tile w (16 output features). K = 128 -> 4 x v_wmma_f32_16x16x32_bf16
// per operand matrix = 8 WMMAs per wave. The mean-divide + bf16 conversion of
// the aggregated messages is fused into the LDS staging (saves a full pass).
__global__ void __launch_bounds__(256)
sage_gemm_wmma_kernel(const float*  __restrict__ msg,  // [N][DIM] f32 edge sums
                      const float*  __restrict__ deg,  // [N] f32 in-degree
                      const __bf16* __restrict__ xin,  // [N][DIM] bf16 self features
                      const __bf16* __restrict__ Wl,   // [DIM][DIM] row-major, bf16
                      const __bf16* __restrict__ Wr,   // [DIM][DIM] row-major, bf16
                      const float*  __restrict__ bias, // [DIM]
                      __bf16* __restrict__ out) {
    __shared__ __bf16 ldsA[16 * DIM];   // aggregated-feature tile (16 x 128)
    __shared__ __bf16 ldsX[16 * DIM];   // self-feature tile       (16 x 128)

    const int mbase = blockIdx.x * 16;
    const int t     = threadIdx.x;

    // cooperative stage: thread t handles 8 consecutive features of row (t>>4);
    // divide by degree and convert f32 -> bf16 on the fly.
    {
        const size_t g  = (size_t)mbase * DIM + t * 8;
        const float  dg = deg[mbase + (t >> 4)];
        const float  rc = (dg > 1.0f) ? (1.0f / dg) : 1.0f;
        v8f  mv = *(const v8f*)&msg[g];
        v8bf av;
#pragma unroll
        for (int i = 0; i < 8; ++i) av[i] = (__bf16)(mv[i] * rc);
        *(v8bf*)&ldsA[t * 8] = av;
        *(v8bf*)&ldsX[t * 8] = *(const v8bf*)&xin[g];
    }
    __syncthreads();

    const int wave = t >> 5;          // N-tile index 0..7
    const int lane = t & 31;
    const int mrow = lane & 15;       // A-fragment row / B-fragment column (mod 16)
    const int half = lane >> 4;       // 0: lanes 0-15, 1: lanes 16-31
    const int ncol = (wave << 4) + mrow;   // output feature owned by this lane

    v8f c = {};

#pragma unroll
    for (int kk = 0; kk < DIM; kk += 32) {
        // A fragment (16x32 bf16): lanes 0-15 hold K {kk..kk+15},
        // lanes 16-31 hold K {kk+16..kk+31}
        const int ka = kk + (half << 4);
        v8bf a0 = *(const v8bf*)&ldsA[mrow * DIM + ka];
        v8bf a1 = *(const v8bf*)&ldsA[mrow * DIM + ka + 8];
        v8bf x0 = *(const v8bf*)&ldsX[mrow * DIM + ka];
        v8bf x1 = *(const v8bf*)&ldsX[mrow * DIM + ka + 8];
        v16bf A = __builtin_shufflevector(a0, a1, 0,1,2,3,4,5,6,7,8,9,10,11,12,13,14,15);
        v16bf X = __builtin_shufflevector(x0, x1, 0,1,2,3,4,5,6,7,8,9,10,11,12,13,14,15);

        // B fragment (32x16 bf16), B[k][n] = W[n][k]: lane's 16 K-values are a
        // contiguous 32-byte slice of W row `ncol`
        const int kb = kk + (half << 4);
        v16bf Bl = *(const v16bf*)&Wl[(size_t)ncol * DIM + kb];
        v16bf Br = *(const v16bf*)&Wr[(size_t)ncol * DIM + kb];

        c = __builtin_amdgcn_wmma_f32_16x16x32_bf16(false, A, false, Bl,
                                                    (short)0, c, false, false);
        c = __builtin_amdgcn_wmma_f32_16x16x32_bf16(false, X, false, Br,
                                                    (short)0, c, false, false);
    }

    // epilogue: bias (indexed by N only), ReLU, convert to bf16 for next layer
    const float bv = bias[ncol];
#pragma unroll
    for (int r = 0; r < 8; ++r) {
        float v = c[r] + bv;
        v = v > 0.0f ? v : 0.0f;
        const int m = mbase + (half << 3) + r;   // D layout: M = r + 8*half
        out[(size_t)m * DIM + ncol] = (__bf16)v;
    }
}

// ---------------- global mean pool ----------------
// batch is sorted, so accumulate run-length segments in registers and flush an
// atomic only when the graph id changes (~32x fewer atomics than per-node).
// One wave per 32 consecutive nodes; lane owns 4 features.
__global__ void pool_scatter_kernel(const __bf16* __restrict__ feat,
                                    const int* __restrict__ batch,
                                    float* __restrict__ gsum,
                                    float* __restrict__ gcnt) {
    int gid   = blockIdx.x * blockDim.x + threadIdx.x;
    int chunk = gid >> 5;
    int lane  = gid & 31;
    int nbase = chunk * 32;
    if (nbase >= N_NODES) return;
    int nend = nbase + 32;
    if (nend > N_NODES) nend = N_NODES;

    float a0 = 0.f, a1 = 0.f, a2 = 0.f, a3 = 0.f, cnt = 0.f;
    int curg = batch[nbase];
    for (int n = nbase; n < nend; ++n) {
        int g = batch[n];                       // uniform across the wave
        if (g != curg) {
            if ((unsigned)curg < N_GRAPHS) {
                float* op = gsum + (size_t)curg * DIM + lane * 4;
                atomic_fadd(op + 0, a0);
                atomic_fadd(op + 1, a1);
                atomic_fadd(op + 2, a2);
                atomic_fadd(op + 3, a3);
                if (lane == 0) atomic_fadd(gcnt + curg, cnt);
            }
            a0 = a1 = a2 = a3 = 0.f; cnt = 0.f; curg = g;
        }
        v4bf v = *(const v4bf*)(feat + (size_t)n * DIM + lane * 4);
        a0 += (float)v[0]; a1 += (float)v[1];
        a2 += (float)v[2]; a3 += (float)v[3];
        cnt += 1.f;
    }
    if ((unsigned)curg < N_GRAPHS) {
        float* op = gsum + (size_t)curg * DIM + lane * 4;
        atomic_fadd(op + 0, a0);
        atomic_fadd(op + 1, a1);
        atomic_fadd(op + 2, a2);
        atomic_fadd(op + 3, a3);
        if (lane == 0) atomic_fadd(gcnt + curg, cnt);
    }
}

__global__ void pool_div_kernel(float* __restrict__ out,
                                const float* __restrict__ cnt) {
    int i = blockIdx.x * blockDim.x + threadIdx.x;
    if (i >= N_GRAPHS * DIM) return;
    float c = cnt[i >> 7];
    out[i] = out[i] / fmaxf(c, 1.0f);
}

// ---------------- launch ----------------
extern "C" void kernel_launch(void* const* d_in, const int* in_sizes, int n_in,
                              void* d_out, int out_size, void* d_ws, size_t ws_size,
                              hipStream_t stream) {
    (void)in_sizes; (void)n_in; (void)out_size; (void)ws_size;

    const float* x     = (const float*)d_in[0];
    const int*   ei    = (const int*)d_in[1];       // [2][N_EDGES]: src row, dst row
    const int*   batch = (const int*)d_in[2];
    const float* Wsrc[6] = { (const float*)d_in[3], (const float*)d_in[4],   // Wl0, Wr0
                             (const float*)d_in[6], (const float*)d_in[7],   // Wl1, Wr1
                             (const float*)d_in[9], (const float*)d_in[10] };// Wl2, Wr2
    const float* bsrc[3] = { (const float*)d_in[5], (const float*)d_in[8],
                             (const float*)d_in[11] };
    const int* src = ei;
    const int* dst = ei + N_EDGES;
    float* out = (float*)d_out;

    // carve workspace (256B aligned chunks), ~103 MB total
    char*  ws  = (char*)d_ws;
    size_t off = 0;
    auto carve = [&](size_t bytes) -> void* {
        void* p = ws + off;
        off += (bytes + 255) & ~(size_t)255;
        return p;
    };
    __bf16* featA = (__bf16*)carve((size_t)N_NODES * DIM * sizeof(__bf16));
    __bf16* featB = (__bf16*)carve((size_t)N_NODES * DIM * sizeof(__bf16));
    float*  msg   = (float*) carve((size_t)N_NODES * DIM * sizeof(float));
    float*  deg   = (float*) carve((size_t)N_NODES * sizeof(float));
    __bf16* wbf   = (__bf16*)carve((size_t)6 * DIM * DIM * sizeof(__bf16));
    float*  gcnt  = (float*) carve((size_t)N_GRAPHS * sizeof(float));

    const int B = 256;
    const int nFeat = N_NODES * DIM;

    // one-time per-call conversions to bf16
    for (int i = 0; i < 6; ++i)
        cvt_f32_bf16_kernel<<<(DIM * DIM + B - 1) / B, B, 0, stream>>>(
            Wsrc[i], wbf + (size_t)i * DIM * DIM, DIM * DIM);
    cvt_f32_bf16_kernel<<<(nFeat + B - 1) / B, B, 0, stream>>>(x, featA, nFeat);

    __bf16* fin  = featA;
    __bf16* fout = featB;
    for (int L = 0; L < 3; ++L) {
        zero_f32_kernel<<<(nFeat + B - 1) / B, B, 0, stream>>>(msg, nFeat);
        zero_f32_kernel<<<(N_NODES + B - 1) / B, B, 0, stream>>>(deg, N_NODES);
        edge_scatter_kernel<<<(N_EDGES * 32 + B - 1) / B, B, 0, stream>>>(
            fin, src, dst, msg, deg);
        sage_gemm_wmma_kernel<<<N_NODES / 16, B, 0, stream>>>(
            msg, deg, fin,
            wbf + (size_t)(2 * L) * DIM * DIM,
            wbf + (size_t)(2 * L + 1) * DIM * DIM,
            bsrc[L], fout);
        __bf16* t = fin; fin = fout; fout = t;
    }

    // global mean pool (fin holds final relu'd features)
    zero_f32_kernel<<<(N_GRAPHS * DIM + B - 1) / B, B, 0, stream>>>(out, N_GRAPHS * DIM);
    zero_f32_kernel<<<(N_GRAPHS + B - 1) / B, B, 0, stream>>>(gcnt, N_GRAPHS);
    pool_scatter_kernel<<<(N_NODES + B - 1) / B, B, 0, stream>>>(fin, batch, out, gcnt);
    pool_div_kernel<<<(N_GRAPHS * DIM + B - 1) / B, B, 0, stream>>>(out, gcnt);
}
